// LoRALinear_60919816126704
// MI455X (gfx1250) — compile-verified, matrix operand order checked by reading, not run on hardware
//
#include <hip/hip_runtime.h>

typedef __bf16 bf16;
typedef __attribute__((ext_vector_type(16))) __bf16 v16bf;
typedef __attribute__((ext_vector_type(8)))  __bf16 v8bf;
typedef __attribute__((ext_vector_type(4)))  __bf16 v4bf;
typedef __attribute__((ext_vector_type(8)))  float  v8f;

#define K_DIM   4096
#define N_DIM   4096
#define M_DIM   8192
#define R_RANK  8
#define SCALING 2.0f

#define XSTR 40   // LDS row stride (bf16 elems): 80B, multiple of 16B
#define WSTR 40

union Frag { v16bf v; v8bf h[2]; };

// ---------------------------------------------------------------------------
// Kernel 1: Weff[o][i] = bf16( W[o][i] + 2 * sum_r A[o][r]*B[i][r] )
// ---------------------------------------------------------------------------
__global__ __launch_bounds__(256) void fold_weff(
    const float* __restrict__ W, const float* __restrict__ A,
    const float* __restrict__ Bm, bf16* __restrict__ Weff)
{
    int idx = blockIdx.x * 256 + threadIdx.x;      // idx = o*K + i
    int o = idx >> 12;
    int i = idx & (K_DIM - 1);
    float acc = W[idx];
#pragma unroll
    for (int r = 0; r < R_RANK; ++r)
        acc += SCALING * A[o * R_RANK + r] * Bm[i * R_RANK + r];
    Weff[idx] = (bf16)acc;
}

// ---------------------------------------------------------------------------
// Pipeline helpers
// ---------------------------------------------------------------------------
__device__ __forceinline__ void wait_async0() {
#if __has_builtin(__builtin_amdgcn_s_wait_asynccnt)
    __builtin_amdgcn_s_wait_asynccnt(0);
#else
    asm volatile("s_wait_asynccnt 0" ::: "memory");
#endif
}

// Issue 4x global_load_b128 of x (f32) into registers; no wait here.
__device__ __forceinline__ void issue_x_loads(const float* __restrict__ x,
                                              int m0, int kk, int tid,
                                              float4* xr) {
#pragma unroll
    for (int p = 0; p < 4; ++p) {
        int linear = (p * 256 + tid) * 4;
        int row = linear >> 5;
        int col = linear & 31;
        xr[p] = *(const float4*)(x + (size_t)(m0 + row) * K_DIM + kk + col);
    }
}

// Convert staged x registers to bf16 and store into LDS (compiler inserts
// the loadcnt wait here, after the WMMA block that sits in between).
__device__ __forceinline__ void store_x_lds(bf16* Xbuf, int tid,
                                            const float4* xr) {
#pragma unroll
    for (int p = 0; p < 4; ++p) {
        int linear = (p * 256 + tid) * 4;
        int row = linear >> 5;
        int col = linear & 31;
        v4bf pk;
        pk[0] = (bf16)xr[p].x; pk[1] = (bf16)xr[p].y;
        pk[2] = (bf16)xr[p].z; pk[3] = (bf16)xr[p].w;
        *(v4bf*)(&Xbuf[row * XSTR + col]) = pk;
    }
}

// Async global->LDS copy of the bf16 Weff tile (2x 16B per lane). Tracked by
// ASYNCcnt; no VGPR round-trip. LDS dest addr = low 32 bits of flat address.
__device__ __forceinline__ void issue_w_async(const bf16* __restrict__ weff,
                                              bf16* Wbuf, int n0, int kk,
                                              int tid) {
#pragma unroll
    for (int p = 0; p < 2; ++p) {
        int linear = (p * 256 + tid) * 8;
        int row = linear >> 5;
        int col = linear & 31;
        const bf16* gsrc = weff + (size_t)(n0 + row) * K_DIM + kk + col;
        unsigned ldsa = (unsigned)(uintptr_t)(&Wbuf[row * WSTR + col]);
        asm volatile("global_load_async_to_lds_b128 %0, %1, off"
                     :: "v"(ldsa), "v"((unsigned long long)(uintptr_t)gsrc)
                     : "memory");
    }
}

// Fragment loads (per ISA 16-bit A/B VGPR layout) + 8 WMMAs.
__device__ __forceinline__ void compute_tile(const bf16* Xbuf, const bf16* Wbuf,
                                             int wm, int wn, int laneN, int laneH,
                                             v8f c[2][4]) {
    Frag a[2], bfr[4];
#pragma unroll
    for (int mt = 0; mt < 2; ++mt) {
        int m = wm * 32 + mt * 16 + laneN;
        a[mt].h[0] = *(const v8bf*)(&Xbuf[m * XSTR + laneH * 8]);
        a[mt].h[1] = *(const v8bf*)(&Xbuf[m * XSTR + 16 + laneH * 8]);
    }
#pragma unroll
    for (int nt = 0; nt < 4; ++nt) {
        int n = wn * 64 + nt * 16 + laneN;
        bfr[nt].h[0] = *(const v8bf*)(&Wbuf[n * WSTR + laneH * 8]);
        bfr[nt].h[1] = *(const v8bf*)(&Wbuf[n * WSTR + 16 + laneH * 8]);
    }
#pragma unroll
    for (int mt = 0; mt < 2; ++mt)
#pragma unroll
        for (int nt = 0; nt < 4; ++nt)
            c[mt][nt] = __builtin_amdgcn_wmma_f32_16x16x32_bf16(
                false, a[mt].v, false, bfr[nt].v,
                (short)0, c[mt][nt], false, false);
}

// ---------------------------------------------------------------------------
// Kernel 2: out[m][n] = sum_k x[m][k] * Weff[n][k] + b[n]
// Double-buffered software pipeline: stage tile k+1 (X: global->regs,
// W: async->LDS) while computing tile k's WMMAs.
// ---------------------------------------------------------------------------
__global__ __launch_bounds__(256) void lora_gemm(
    const float* __restrict__ x, const bf16* __restrict__ weff,
    const float* __restrict__ bias, float* __restrict__ out)
{
    __shared__ __align__(16) bf16 Xs[2][128 * XSTR];
    __shared__ __align__(16) bf16 Ws[2][128 * WSTR];

    const int tid   = threadIdx.x;
    const int lane  = tid & 31;
    const int w     = tid >> 5;
    const int wm    = w & 3;         // 4 waves along M (32 rows each)
    const int wn    = w >> 2;        // 2 waves along N (64 cols each)
    const int m0    = blockIdx.y * 128;
    const int n0    = blockIdx.x * 128;
    const int laneN = lane & 15;
    const int laneH = lane >> 4;

    v8f c[2][4];
#pragma unroll
    for (int mt = 0; mt < 2; ++mt)
#pragma unroll
        for (int nt = 0; nt < 4; ++nt)
            c[mt][nt] = (v8f){};

    float4 xr[4];

    // ---- prologue: stage tile 0 into buffer 0 ----
    issue_x_loads(x, m0, 0, tid, xr);
    issue_w_async(weff, Ws[0], n0, 0, tid);
    store_x_lds(Xs[0], tid, xr);
    wait_async0();
    __syncthreads();

    int ibuf = 0;
    for (int kk = 0; kk < K_DIM - 32; kk += 32) {
        const int nbuf = ibuf ^ 1;
        // issue next tile's staging (overlaps with this tile's WMMAs)
        issue_x_loads(x, m0, kk + 32, tid, xr);
        issue_w_async(weff, Ws[nbuf], n0, kk + 32, tid);
        // compute current tile
        compute_tile(Xs[ibuf], Ws[ibuf], wm, wn, laneN, laneH, c);
        // finish next tile's staging
        store_x_lds(Xs[nbuf], tid, xr);
        wait_async0();
        __syncthreads();
        ibuf = nbuf;
    }
    // final tile
    compute_tile(Xs[ibuf], Ws[ibuf], wm, wn, laneN, laneH, c);

    // ---- epilogue: bias add + store (C/D layout: VGPR r -> M = r + 8*laneH)
#pragma unroll
    for (int nt = 0; nt < 4; ++nt) {
        int n = n0 + wn * 64 + nt * 16 + laneN;
        float bv = bias[n];
#pragma unroll
        for (int mt = 0; mt < 2; ++mt) {
            int mb = m0 + wm * 32 + mt * 16 + laneH * 8;
#pragma unroll
            for (int r = 0; r < 8; ++r)
                out[(size_t)(mb + r) * N_DIM + n] = c[mt][nt][r] + bv;
        }
    }
}

// ---------------------------------------------------------------------------
extern "C" void kernel_launch(void* const* d_in, const int* in_sizes, int n_in,
                              void* d_out, int out_size, void* d_ws, size_t ws_size,
                              hipStream_t stream) {
    const float* x  = (const float*)d_in[0];
    const float* W  = (const float*)d_in[1];
    const float* b  = (const float*)d_in[2];
    const float* A  = (const float*)d_in[3];
    const float* Bm = (const float*)d_in[4];
    float* out = (float*)d_out;
    bf16* weff = (bf16*)d_ws;   // 4096*4096*2 = 32 MB scratch

    fold_weff<<<(N_DIM * K_DIM) / 256, 256, 0, stream>>>(W, A, Bm, weff);

    dim3 grid(N_DIM / 128, M_DIM / 128);
    lora_gemm<<<grid, 256, 0, stream>>>(x, weff, b, out);
}
